// TransitionDownWithDistFea_23519240913433
// MI455X (gfx1250) — matrix operand order repaired and checked by reading, not compiled, not run on hardware
//
#include <hip/hip_runtime.h>
#include <hip/hip_bf16.h>

typedef __attribute__((ext_vector_type(2)))  float    v2f;
typedef __attribute__((ext_vector_type(8)))  float    v8f;
typedef __attribute__((ext_vector_type(16))) _Float16 v16h;

#define KNN 16
#define NPTS 8192
#define SPTS 4096

// per-point: squared norm + padded (x,y,z,0) fragment source for WMMA loads
__global__ void pack_kernel(const float* __restrict__ xyz, float* __restrict__ sq,
                            float* __restrict__ pad, int n) {
    int i = blockIdx.x * blockDim.x + threadIdx.x;
    if (i >= n) return;
    const float* p = xyz + (size_t)i * 3;
    float x = p[0], y = p[1], z = p[2];
    sq[i] = x * x + y * y + z * z;
    float* o = pad + (size_t)i * 4;
    o[0] = x; o[1] = y; o[2] = z; o[3] = 0.0f;
}

__device__ __forceinline__ float pdist(const float* __restrict__ xyz, const float* __restrict__ sq,
                                       int i, int j) {
    const float* a = xyz + (size_t)i * 3;
    const float* b = xyz + (size_t)j * 3;
    float dot = a[0]*b[0] + a[1]*b[1] + a[2]*b[2];
    float d2  = sq[i] + sq[j] - 2.0f * dot;
    d2 = fmaxf(d2, 0.0f);
    return (d2 > 0.0f) ? __builtin_amdgcn_sqrtf(d2) : 0.0f;
}

// One wave per 16-row tile. Gram tile via V_WMMA_F32_16X16X4_F32 on padded
// (x,y,z,0) fragments (branch-free b64 loads). Top-16 kept on squared
// distances (monotone in d); sqrt applied only to the 16 winners per row.
__global__ __launch_bounds__(32) void topk_kernel(const float* __restrict__ pad,
                                                  const float* __restrict__ sq,
                                                  int n, int* __restrict__ ni,
                                                  float* __restrict__ nd) {
    __shared__ float tile[16 * 16];
    __shared__ float md[32 * KNN];
    __shared__ int   mi[32 * KNN];

    const int lane = threadIdx.x;
    const int hi   = lane >> 4;      // 0: holds (x,y)=K0,1   1: holds (z,0)=K2,3
    const int ln   = lane & 15;
    const int r0   = blockIdx.x * 16;

    v2f afrag = *(const v2f*)(pad + (size_t)(r0 + ln) * 4 + hi * 2);
    float sqA[8];
#pragma unroll
    for (int v = 0; v < 8; ++v) sqA[v] = sq[r0 + v + hi * 8];

    float bd[KNN];
    int   bi[KNN];
#pragma unroll
    for (int t = 0; t < KNN; ++t) { bd[t] = 1e30f; bi[t] = 0x7fffffff; }

    for (int c0 = 0; c0 < n; c0 += 16) {
        v2f bfrag = *(const v2f*)(pad + (size_t)(c0 + ln) * 4 + hi * 2);
        float sqB = sq[c0 + ln];

        v8f c = {};
        c = __builtin_amdgcn_wmma_f32_16x16x4_f32(false, afrag, false, bfrag,
                                                  (short)0, c, false, false);
        __syncthreads();
#pragma unroll
        for (int v = 0; v < 8; ++v) {
            float d2 = fmaxf(sqA[v] + sqB - 2.0f * c[v], 0.0f);
            tile[(v + hi * 8) * 16 + ln] = d2;
        }
        __syncthreads();
        // read phase: this lane owns row ln, columns [hi*8, hi*8+8)
#pragma unroll
        for (int j = 0; j < 8; ++j) {
            float d  = tile[ln * 16 + hi * 8 + j];
            int  idx = c0 + hi * 8 + j;
            if (d < bd[KNN - 1] || (d == bd[KNN - 1] && idx < bi[KNN - 1])) {
                bd[KNN - 1] = d; bi[KNN - 1] = idx;
#pragma unroll
                for (int t = KNN - 1; t > 0; --t) {
                    bool sw = (bd[t] < bd[t - 1]) ||
                              (bd[t] == bd[t - 1] && bi[t] < bi[t - 1]);
                    if (sw) {
                        float td = bd[t]; bd[t] = bd[t - 1]; bd[t - 1] = td;
                        int   ti = bi[t]; bi[t] = bi[t - 1]; bi[t - 1] = ti;
                    }
                }
            }
        }
    }

    // merge the two sorted half-lists per row; sqrt only the winners
#pragma unroll
    for (int t = 0; t < KNN; ++t) { md[lane * KNN + t] = bd[t]; mi[lane * KNN + t] = bi[t]; }
    __syncthreads();
    if (lane < 16) {
        int pa = lane * KNN, pb = (lane + 16) * KNN;
        int ia = 0, ib = 0;
        int row = r0 + lane;
#pragma unroll
        for (int t = 0; t < KNN; ++t) {
            float da = md[pa + ia], db = md[pb + ib];
            int   xa = mi[pa + ia], xb = mi[pb + ib];
            bool takeA = (da < db) || (da == db && xa <= xb);
            float d2 = takeA ? da : db;
            nd[(size_t)row * KNN + t] = (d2 > 0.0f) ? __builtin_amdgcn_sqrtf(d2) : 0.0f;
            ni[(size_t)row * KNN + t] = takeA ? xa : xb;
            if (takeA) ++ia; else ++ib;
        }
    }
}

// intra features: nd[1..3] plus pairwise distances among neighbors 1,2,3
__global__ void intra_kernel(const float* __restrict__ xyz, const float* __restrict__ sq,
                             const int* __restrict__ ni, const float* __restrict__ nd,
                             float* __restrict__ intra, int n) {
    int i = blockIdx.x * blockDim.x + threadIdx.x;
    if (i >= n) return;
    int a1 = ni[(size_t)i * KNN + 1];
    int a2 = ni[(size_t)i * KNN + 2];
    int a3 = ni[(size_t)i * KNN + 3];
    float* o = intra + (size_t)i * 6;
    o[0] = nd[(size_t)i * KNN + 1];
    o[1] = nd[(size_t)i * KNN + 2];
    o[2] = nd[(size_t)i * KNN + 3];
    o[3] = pdist(xyz, sq, a1, a2);
    o[4] = pdist(xyz, sq, a1, a3);
    o[5] = pdist(xyz, sq, a2, a3);
}

// gather xyz_s, feat_s, intra_before_s via sample_indexes
__global__ void gather_kernel(const float* __restrict__ xyz, const float* __restrict__ feature,
                              const int* __restrict__ samp, const float* __restrict__ intra_b,
                              float* __restrict__ xyz_s, float* __restrict__ feat_s,
                              float* __restrict__ intra_bs, int S) {
    int i = blockIdx.x * blockDim.x + threadIdx.x;
    if (i >= S) return;
    int src = samp[i];
#pragma unroll
    for (int c = 0; c < 3; ++c)  xyz_s[(size_t)i * 3 + c]  = xyz[(size_t)src * 3 + c];
    for (int c = 0; c < 64; ++c) feat_s[(size_t)i * 64 + c] = feature[(size_t)src * 64 + c];
#pragma unroll
    for (int c = 0; c < 6; ++c)  intra_bs[(size_t)i * 6 + c] = intra_b[(size_t)src * 6 + c];
}

// rf_after (S x K x 28) + ni_after as float
__global__ void rf_after_kernel(const float* __restrict__ xyz_s, const float* __restrict__ sq_a,
                                const int* __restrict__ ni_a, const float* __restrict__ intra_a,
                                float* __restrict__ rf_out, float* __restrict__ ni_out, int S) {
    int t = blockIdx.x * blockDim.x + threadIdx.x;
    if (t >= S * KNN) return;
    int n = t / KNN;
    int nk = ni_a[t];
    float* o = rf_out + (size_t)t * 28;
#pragma unroll
    for (int c = 0; c < 6; ++c) o[c]     = intra_a[(size_t)n * 6 + c];
#pragma unroll
    for (int c = 0; c < 6; ++c) o[6 + c] = intra_a[(size_t)nk * 6 + c];
#pragma unroll
    for (int a1 = 0; a1 < 4; ++a1) {
        int i1 = ni_a[(size_t)n * KNN + a1];
#pragma unroll
        for (int a2 = 0; a2 < 4; ++a2) {
            int i2 = ni_a[(size_t)nk * KNN + a2];
            o[12 + a1 * 4 + a2] = pdist(xyz_s, sq_a, i1, i2);
        }
    }
    ni_out[t] = (float)nk;
}

// adf (S x 28): intra_before_s | intra_after | inter distances in before-space
__global__ void adf_kernel(const float* __restrict__ xyz_b, const float* __restrict__ sq_b,
                           const int* __restrict__ ni_b, const int* __restrict__ ni_a,
                           const int* __restrict__ samp, const float* __restrict__ intra_bs,
                           const float* __restrict__ intra_a, float* __restrict__ adf, int S) {
    int n = blockIdx.x * blockDim.x + threadIdx.x;
    if (n >= S) return;
    float* o = adf + (size_t)n * 28;
#pragma unroll
    for (int c = 0; c < 6; ++c) { o[c] = intra_bs[(size_t)n * 6 + c]; o[6 + c] = intra_a[(size_t)n * 6 + c]; }
    int sb = samp[n];
#pragma unroll
    for (int a1 = 0; a1 < 4; ++a1) {
        int i1 = ni_b[(size_t)sb * KNN + a1];
#pragma unroll
        for (int a2 = 0; a2 < 4; ++a2) {
            int i2 = samp[ni_a[(size_t)n * KNN + a2]];
            o[12 + a1 * 4 + a2] = pdist(xyz_b, sq_b, i1, i2);
        }
    }
}

// Y = X @ W + bias via V_WMMA_F32_16X16X32_F16; one wave per 16x16 tile.
__global__ __launch_bounds__(32) void gemm_kernel(const float* __restrict__ X,
                                                  const float* __restrict__ W,
                                                  const float* __restrict__ bias,
                                                  float* __restrict__ Y,
                                                  int S, int Cin, int Cout) {
    const int lane = threadIdx.x;
    const int hi   = lane >> 4;
    const int ln   = lane & 15;
    const int m0   = blockIdx.x * 16;
    const int n0   = blockIdx.y * 16;
    const int row  = m0 + ln;   // A-matrix row (same for both lane halves)
    const int col  = n0 + ln;   // B-matrix column

    v8f acc = {};
    for (int k0 = 0; k0 < Cin; k0 += 32) {
        v16h a, b;
#pragma unroll
        for (int h = 0; h < 16; ++h) {
            // A k order used here: lanes<16 K=0..15, lanes>=16 K=16..31 (contiguous, same as B)
            int ka = k0 + (hi ? 16 : 0) + h;
            a[h] = (ka < Cin) ? (_Float16)X[(size_t)row * Cin + ka] : (_Float16)0.0f;
            // B k order used here: lanes<16 K=0..15, lanes>=16 K=16..31 (contiguous, same as A)
            int kb = k0 + (hi ? 16 : 0) + h;
            b[h] = (kb < Cin) ? (_Float16)W[(size_t)kb * Cout + col] : (_Float16)0.0f;
        }
        acc = __builtin_amdgcn_wmma_f32_16x16x32_f16(false, a, false, b,
                                                     (short)0, acc, false, false);
    }
#pragma unroll
    for (int v = 0; v < 8; ++v) {
        int r = m0 + v + hi * 8;
        Y[(size_t)r * Cout + col] = acc[v] + bias[col];
    }
}

// per-channel batchnorm over S rows (g*xhat + beta), optional leaky-relu(0.2)
__global__ __launch_bounds__(256) void bn_kernel(float* __restrict__ Y,
                                                 const float* __restrict__ g,
                                                 const float* __restrict__ be,
                                                 int S, int C, int leaky) {
    __shared__ float ssum[256];
    __shared__ float ssq[256];
    const int c   = blockIdx.x;
    const int tid = threadIdx.x;
    float s = 0.0f, s2 = 0.0f;
    for (int r = tid; r < S; r += 256) {
        float y = Y[(size_t)r * C + c];
        s += y; s2 += y * y;
    }
    ssum[tid] = s; ssq[tid] = s2;
    __syncthreads();
    for (int off = 128; off > 0; off >>= 1) {
        if (tid < off) { ssum[tid] += ssum[tid + off]; ssq[tid] += ssq[tid + off]; }
        __syncthreads();
    }
    float mu  = ssum[0] / (float)S;
    float var = ssq[0] / (float)S - mu * mu;
    float inv = rsqrtf(var + 1e-5f);
    float gg = g[c], bb = be[c];
    for (int r = tid; r < S; r += 256) {
        float y = (Y[(size_t)r * C + c] - mu) * inv * gg + bb;
        if (leaky) y = (y > 0.0f) ? y : 0.2f * y;
        Y[(size_t)r * C + c] = y;
    }
}

// feat2[:, :64] = leaky(feat_s * w + bia); feat2[:, 64:92] = adf
__global__ void fuse_kernel(const float* __restrict__ feat_s, const float* __restrict__ w,
                            const float* __restrict__ bia, const float* __restrict__ adf,
                            float* __restrict__ feat2, int S) {
    int t = blockIdx.x * blockDim.x + threadIdx.x;
    if (t >= S * 92) return;
    int n = t / 92, c = t % 92;
    float v;
    if (c < 64) {
        size_t k = (size_t)n * 64 + c;
        float y = feat_s[k] * w[k] + bia[k];
        v = (y > 0.0f) ? y : 0.2f * y;
    } else {
        v = adf[(size_t)n * 28 + (c - 64)];
    }
    feat2[t] = v;
}

extern "C" void kernel_launch(void* const* d_in, const int* in_sizes, int n_in,
                              void* d_out, int out_size, void* d_ws, size_t ws_size,
                              hipStream_t stream) {
    (void)in_sizes; (void)n_in; (void)out_size; (void)ws_size;
    const int N = NPTS, S = SPTS;

    const float* xyz     = (const float*)d_in[0];
    const float* feature = (const float*)d_in[1];
    const int*   samp    = (const int*)  d_in[2];
    const float* Ww = (const float*)d_in[3];
    const float* bw = (const float*)d_in[4];
    const float* gw = (const float*)d_in[5];
    const float* betaw = (const float*)d_in[6];
    const float* Wb = (const float*)d_in[7];
    const float* bb = (const float*)d_in[8];
    const float* gb = (const float*)d_in[9];
    const float* betab = (const float*)d_in[10];
    const float* Wo = (const float*)d_in[11];
    const float* bo = (const float*)d_in[12];
    const float* go = (const float*)d_in[13];
    const float* betao = (const float*)d_in[14];

    float* out      = (float*)d_out;
    float* xyz_s    = out;                               // 4096*3
    float* feat_out = out + (size_t)S * 3;               // 4096*128
    float* rf_out   = feat_out + (size_t)S * 128;        // 4096*16*28
    float* ni_out   = rf_out + (size_t)S * KNN * 28;     // 4096*16

    char*  wsb = (char*)d_ws;
    size_t off = 0;
    auto allocF = [&](size_t cnt) { float* p = (float*)(wsb + off); off += cnt * 4; return p; };
    float* sq_b     = allocF(N);
    float* pad_b    = allocF((size_t)N * 4);
    int*   ni_b     = (int*)allocF((size_t)N * KNN);
    float* nd_b     = allocF((size_t)N * KNN);
    float* intra_b  = allocF((size_t)N * 6);
    float* sq_a     = allocF(S);
    float* pad_a    = allocF((size_t)S * 4);
    int*   ni_a     = (int*)allocF((size_t)S * KNN);
    float* nd_a     = allocF((size_t)S * KNN);
    float* intra_a  = allocF((size_t)S * 6);
    float* feat_s   = allocF((size_t)S * 64);
    float* intra_bs = allocF((size_t)S * 6);
    float* adf      = allocF((size_t)S * 28);
    float* wbuf     = allocF((size_t)S * 64);
    float* bbuf     = allocF((size_t)S * 64);
    float* feat2    = allocF((size_t)S * 92);

    // generator(xyz) -> ni_b, nd_b, intra_b
    pack_kernel<<<N / 256, 256, 0, stream>>>(xyz, sq_b, pad_b, N);
    topk_kernel<<<N / 16, 32, 0, stream>>>(pad_b, sq_b, N, ni_b, nd_b);
    intra_kernel<<<N / 256, 256, 0, stream>>>(xyz, sq_b, ni_b, nd_b, intra_b, N);

    // subsample
    gather_kernel<<<S / 256, 256, 0, stream>>>(xyz, feature, samp, intra_b,
                                               xyz_s, feat_s, intra_bs, S);

    // generator(xyz_s) -> ni_a, nd_a, intra_a
    pack_kernel<<<S / 256, 256, 0, stream>>>(xyz_s, sq_a, pad_a, S);
    topk_kernel<<<S / 16, 32, 0, stream>>>(pad_a, sq_a, S, ni_a, nd_a);
    intra_kernel<<<S / 256, 256, 0, stream>>>(xyz_s, sq_a, ni_a, nd_a, intra_a, S);

    // rf_after + ni_after outputs
    rf_after_kernel<<<(S * KNN) / 256, 256, 0, stream>>>(xyz_s, sq_a, ni_a, intra_a,
                                                         rf_out, ni_out, S);
    // adf descriptor
    adf_kernel<<<S / 256, 256, 0, stream>>>(xyz, sq_b, ni_b, ni_a, samp,
                                            intra_bs, intra_a, adf, S);

    // w = BN(adf @ Ww + bw); bia = BN(adf @ Wb + bb)
    gemm_kernel<<<dim3(S / 16, 64 / 16), 32, 0, stream>>>(adf, Ww, bw, wbuf, S, 28, 64);
    bn_kernel<<<64, 256, 0, stream>>>(wbuf, gw, betaw, S, 64, 0);
    gemm_kernel<<<dim3(S / 16, 64 / 16), 32, 0, stream>>>(adf, Wb, bb, bbuf, S, 28, 64);
    bn_kernel<<<64, 256, 0, stream>>>(bbuf, gb, betab, S, 64, 0);

    // feat2 = [leaky(feat*w + bia), adf]
    fuse_kernel<<<(S * 92 + 255) / 256, 256, 0, stream>>>(feat_s, wbuf, bbuf, adf, feat2, S);

    // feat_out = leaky(BN(feat2 @ Wo + bo))
    gemm_kernel<<<dim3(S / 16, 128 / 16), 32, 0, stream>>>(feat2, Wo, bo, feat_out, S, 92, 128);
    bn_kernel<<<128, 256, 0, stream>>>(feat_out, go, betao, S, 128, 1);
}
